// GlobalWorkSpaceRNNofRNNs_67388036874680
// MI455X (gfx1250) — matrix operand order repaired and checked by reading, not compiled, not run on hardware
//
#include <hip/hip_runtime.h>
#include <hip/hip_bf16.h>
#include <math.h>

// ---------------- problem constants ----------------
#define ALPHA_C 0.1f
#define B_SZ 64
#define T_LEN 256
#define IN_D  512
#define H_D   2048
#define OUT_D 512

typedef __bf16 bf16_t;
typedef __attribute__((ext_vector_type(16))) __bf16 v16bf;
typedef __attribute__((ext_vector_type(8)))  __bf16 v8bf;
typedef __attribute__((ext_vector_type(8)))  float  v8f;

union V16U { v16bf v; v8bf h[2]; };

// ---------------- WMMA helpers (wave32, 16x16x32 bf16) ----------------

// B operand from a pre-swizzled tile buffer: tile = 32 lanes x 16 bf16 contiguous.
__device__ __forceinline__ v16bf load_B_sw(const bf16_t* tiles, int tileIdx) {
    int lane = threadIdx.x & 31;
    const bf16_t* p = tiles + ((size_t)tileIdx * 32 + lane) * 16;
    V16U u;
    u.h[0] = *(const v8bf*)(p);
    u.h[1] = *(const v8bf*)(p + 8);
    return u.v;
}

__device__ __forceinline__ v8f wmma_bf16(v16bf a, v16bf b, v8f c) {
    return __builtin_amdgcn_wmma_f32_16x16x32_bf16(
        /*neg_a=*/false, a, /*neg_b=*/false, b,
        /*c_mod=*/(short)0, c, /*reuse_a=*/false, /*reuse_b=*/false);
}

// LDS-staged GEMM strip: block-cooperatively copies a contiguous 16xKDIM bf16 A
// strip (row-major, ld == KDIM) into LDS, then each wave runs the K loop with
// A from LDS (ds_load_b128 x2 per tile) and pre-swizzled B from global
// (global_load_b128 x2 per tile), accumulating one 16x16 f32 tile.
// Wave32 A layout: lanes 0-15 -> rows M=lane, K runs {k0..k0+15};
// lanes 16-31 -> rows M=lane-16, K runs {k0+16..k0+31}.
template <int KDIM>
__device__ __forceinline__ v8f gemm_strip_lds(const bf16_t* __restrict__ Aglob,
                                              const bf16_t* __restrict__ Bnt,
                                              bf16_t* lds) {
    const int tid = threadIdx.x;
    constexpr int KT     = KDIM / 32;             // K tiles
    constexpr int CHUNKS = (16 * KDIM) / (8 * 256); // b128 chunks per thread
    #pragma unroll
    for (int c = 0; c < CHUNKS; ++c) {
        int idx = (c * 256 + tid) * 8;
        *(v8bf*)(lds + idx) = *(const v8bf*)(Aglob + idx);
    }
    __syncthreads();

    const int lane = tid & 31;
    const int r    = lane & 15;
    const int hi   = (lane >> 4) & 1;
    const bf16_t* arow = lds + r * KDIM + hi * 16;

    v8f acc = {};
    #pragma unroll 4
    for (int kt = 0; kt < KT; ++kt) {
        __builtin_prefetch(Bnt + (size_t)(kt + 2) * 512, 0, 1); // next B tiles -> cache
        V16U a;
        a.h[0] = *(const v8bf*)(arow + kt * 32);
        a.h[1] = *(const v8bf*)(arow + kt * 32 + 8);
        v16bf b = load_B_sw(Bnt, kt);
        acc = wmma_bf16(a.v, b, acc);
    }
    return acc;
}

// B-swizzle K mapping: element e of v16 holds K = k0 + e + (lane>=16 ? 16 : 0), N = lane&15.

// ---------------- prep kernels ----------------

// W_hat[j][k] = Bm[j][k] - m[j] * Bm[k][j] / m[k], stored as swizzled B-operand tiles.
// Tile grid: nt in [0,128) over j, kt in [0,64) over k (K-tile width 32).
__global__ void prep_whatB(const float* __restrict__ B, const float* __restrict__ Bmask,
                           const float* __restrict__ m, bf16_t* __restrict__ whatB) {
    int tid  = blockIdx.x * blockDim.x + threadIdx.x;
    int lane = tid & 31;
    int tile = tid >> 5;
    if (tile >= 128 * 64) return;
    int nt = tile / 64, kt = tile % 64;
    int hi = (lane >> 4) & 1;
    int j  = nt * 16 + (lane & 15);
    float mj = m[j];
    bf16_t* dst = whatB + ((size_t)tile * 32 + lane) * 16;
    #pragma unroll
    for (int e = 0; e < 16; ++e) {
        int k = kt * 32 + e + hi * 16;
        float bm_jk = B[(size_t)j * H_D + k] * Bmask[(size_t)j * H_D + k];
        float bm_kj = B[(size_t)k * H_D + j] * Bmask[(size_t)k * H_D + j];
        float w = bm_jk - mj * bm_kj / m[k];
        dst[e] = (bf16_t)w;
    }
}

// W_in as B-operand: B[k][n] = W_in[n][k].  nt in [0,128), kt in [0,16).
__global__ void prep_winB(const float* __restrict__ Win, bf16_t* __restrict__ winB) {
    int tid  = blockIdx.x * blockDim.x + threadIdx.x;
    int lane = tid & 31;
    int tile = tid >> 5;
    if (tile >= 128 * 16) return;
    int nt = tile / 16, kt = tile % 16;
    int hi = (lane >> 4) & 1;
    int n  = nt * 16 + (lane & 15);
    bf16_t* dst = winB + ((size_t)tile * 32 + lane) * 16;
    #pragma unroll
    for (int e = 0; e < 16; ++e) {
        int k = kt * 32 + e + hi * 16;
        dst[e] = (bf16_t)Win[(size_t)n * IN_D + k];
    }
}

// W_out as B-operand: B[k][n] = W_out[n][k].  nt in [0,32), kt in [0,64).
__global__ void prep_woutB(const float* __restrict__ Wout, bf16_t* __restrict__ woutB) {
    int tid  = blockIdx.x * blockDim.x + threadIdx.x;
    int lane = tid & 31;
    int tile = tid >> 5;
    if (tile >= 32 * 64) return;
    int nt = tile / 64, kt = tile % 64;
    int hi = (lane >> 4) & 1;
    int n  = nt * 16 + (lane & 15);
    bf16_t* dst = woutB + ((size_t)tile * 32 + lane) * 16;
    #pragma unroll
    for (int e = 0; e < 16; ++e) {
        int k = kt * 32 + e + hi * 16;
        dst[e] = (bf16_t)Wout[(size_t)n * H_D + k];
    }
}

// input [B,T,IN] fp32 -> inb [T*B, IN] bf16 with row m = t*64 + b.
__global__ void conv_input(const float* __restrict__ inp, bf16_t* __restrict__ inb) {
    size_t tid = (size_t)blockIdx.x * blockDim.x + threadIdx.x;
    if (tid >= (size_t)T_LEN * B_SZ * IN_D) return;
    int i = (int)(tid % IN_D);
    int m = (int)(tid / IN_D);
    int t = m / B_SZ, b = m % B_SZ;
    inb[tid] = (bf16_t)inp[((size_t)b * T_LEN + t) * IN_D + i];
}

// zero h_all slot 0 (bf16) and h_fp32 buffer 0.
__global__ void zero_init(bf16_t* __restrict__ h_all0, float* __restrict__ h0) {
    int tid = blockIdx.x * blockDim.x + threadIdx.x;
    if (tid >= B_SZ * H_D) return;
    h_all0[tid] = (bf16_t)0.0f;
    h0[tid] = 0.0f;
}

// ---------------- GEMM kernels ----------------
// All GEMM kernels: 256 threads = 8 waves; all 8 waves of a block share the
// same mt (A strip), staged once in LDS; grids are exact (no guards).

// x_proj = inb @ W_in^T + b_h   -> fp32 [T*B, H]. One wave per 16x16 tile.
__global__ void xproj_gemm(const bf16_t* __restrict__ inb, const bf16_t* __restrict__ winB,
                           const float* __restrict__ bh, float* __restrict__ xproj) {
    __shared__ __align__(16) bf16_t lds[16 * IN_D];   // 16 KiB
    int w  = blockIdx.x * 8 + (threadIdx.x >> 5);
    int mt = w / 128, nt = w % 128;
    int lane = threadIdx.x & 31;
    int hi = (lane >> 4) & 1;

    v8f acc = gemm_strip_lds<IN_D>(inb + (size_t)mt * 16 * IN_D,
                                   winB + (size_t)nt * 16 * 512, lds);

    int n = nt * 16 + (lane & 15);
    float bias = bh[n];
    #pragma unroll
    for (int r = 0; r < 8; ++r) {
        int m = mt * 16 + r + hi * 8;
        xproj[(size_t)m * H_D + n] = acc[r] + bias;
    }
}

// One Euler step: acc = h_prev @ W_hat^T; h_new = h + ALPHA*(tanh(xp + acc) - h).
__global__ void step_kernel(const bf16_t* __restrict__ hA, const bf16_t* __restrict__ whatB,
                            const float* __restrict__ xproj_t,
                            const float* __restrict__ hprev, float* __restrict__ hnext,
                            bf16_t* __restrict__ hA_next) {
    __shared__ __align__(16) bf16_t lds[16 * H_D];    // 64 KiB
    int w  = blockIdx.x * 8 + (threadIdx.x >> 5);
    int mt = w / 128, nt = w % 128;
    int lane = threadIdx.x & 31;
    int hi = (lane >> 4) & 1;

    v8f acc = gemm_strip_lds<H_D>(hA + (size_t)mt * 16 * H_D,
                                  whatB + (size_t)nt * 64 * 512, lds);

    int n = nt * 16 + (lane & 15);
    #pragma unroll
    for (int r = 0; r < 8; ++r) {
        int b = mt * 16 + r + hi * 8;            // batch row (M = 64)
        size_t idx = (size_t)b * H_D + n;
        float hp = hprev[idx];
        float f  = tanhf(xproj_t[idx] + acc[r]);
        float hn = hp + ALPHA_C * (f - hp);
        hnext[idx]   = hn;
        hA_next[idx] = (bf16_t)hn;
    }
}

// out[b][t][o] = h_all[t+1][b] @ W_out^T + b_out. A rows m = t*64+b at h_all + 64*H.
__global__ void out_gemm(const bf16_t* __restrict__ hAll1, const bf16_t* __restrict__ woutB,
                         const float* __restrict__ bout, float* __restrict__ out) {
    __shared__ __align__(16) bf16_t lds[16 * H_D];    // 64 KiB
    int w  = blockIdx.x * 8 + (threadIdx.x >> 5);
    int mt = w / 32, nt = w % 32;
    int lane = threadIdx.x & 31;
    int hi = (lane >> 4) & 1;

    v8f acc = gemm_strip_lds<H_D>(hAll1 + (size_t)mt * 16 * H_D,
                                  woutB + (size_t)nt * 64 * 512, lds);

    int o = nt * 16 + (lane & 15);
    float bias = bout[o];
    #pragma unroll
    for (int r = 0; r < 8; ++r) {
        int m = mt * 16 + r + hi * 8;            // m = t*64 + b
        int t = m >> 6, b = m & 63;
        out[((size_t)b * T_LEN + t) * OUT_D + o] = acc[r] + bias;
    }
}

// ---------------- host launcher ----------------

extern "C" void kernel_launch(void* const* d_in, const int* in_sizes, int n_in,
                              void* d_out, int out_size, void* d_ws, size_t ws_size,
                              hipStream_t stream) {
    const float* inp   = (const float*)d_in[0];
    const float* Bmat  = (const float*)d_in[1];
    const float* Bmask = (const float*)d_in[2];
    const float* mdiag = (const float*)d_in[3];
    const float* Win   = (const float*)d_in[4];
    const float* bh    = (const float*)d_in[5];
    const float* Wout  = (const float*)d_in[6];
    const float* bout  = (const float*)d_in[7];
    float* out = (float*)d_out;

    // workspace layout (bytes, 256-aligned sections)
    char* w = (char*)d_ws;
    const size_t SZ_WHAT  = (size_t)H_D * H_D * 2;                 //   8 MiB
    const size_t SZ_WIN   = (size_t)H_D * IN_D * 2;                //   2 MiB
    const size_t SZ_WOUT  = (size_t)OUT_D * H_D * 2;               //   2 MiB
    const size_t SZ_INB   = (size_t)T_LEN * B_SZ * IN_D * 2;       //  16 MiB
    const size_t SZ_XPROJ = (size_t)T_LEN * B_SZ * H_D * 4;        // 128 MiB
    const size_t SZ_HF    = (size_t)B_SZ * H_D * 4;                // 512 KiB each
    const size_t SLOT     = (size_t)B_SZ * H_D;                    // elements per h slot

    size_t off = 0;
    bf16_t* whatB = (bf16_t*)(w + off); off += SZ_WHAT;
    bf16_t* winB  = (bf16_t*)(w + off); off += SZ_WIN;
    bf16_t* woutB = (bf16_t*)(w + off); off += SZ_WOUT;
    bf16_t* inb   = (bf16_t*)(w + off); off += SZ_INB;
    float*  xproj = (float*) (w + off); off += SZ_XPROJ;
    float*  hf0   = (float*) (w + off); off += SZ_HF;
    float*  hf1   = (float*) (w + off); off += SZ_HF;
    bf16_t* h_all = (bf16_t*)(w + off);                            // (T+1) slots, 64 MiB

    // --- prep ---
    prep_whatB<<<(128 * 64 * 32) / 256, 256, 0, stream>>>(Bmat, Bmask, mdiag, whatB);
    prep_winB <<<(128 * 16 * 32) / 256, 256, 0, stream>>>(Win, winB);
    prep_woutB<<<(32 * 64 * 32) / 256, 256, 0, stream>>>(Wout, woutB);
    conv_input<<<((size_t)T_LEN * B_SZ * IN_D + 255) / 256, 256, 0, stream>>>(inp, inb);
    zero_init <<<(B_SZ * H_D + 255) / 256, 256, 0, stream>>>(h_all, hf0);

    // --- x_proj GEMM: 1024 mt x 128 nt tiles, 8 waves/block ---
    xproj_gemm<<<131072 / 8, 256, 0, stream>>>(inb, winB, bh, xproj);

    // --- sequential scan: one kernel per step (kernel boundary = global sync) ---
    for (int t = 0; t < T_LEN; ++t) {
        const bf16_t* hA   = h_all + (size_t)t * SLOT;
        bf16_t* hA_next    = h_all + (size_t)(t + 1) * SLOT;
        const float* xp_t  = xproj + (size_t)t * SLOT;
        const float* hprev = (t & 1) ? hf1 : hf0;
        float* hnext       = (t & 1) ? hf0 : hf1;
        step_kernel<<<512 / 8, 256, 0, stream>>>(hA, whatB, xp_t, hprev, hnext, hA_next);
    }

    // --- output projection over all timesteps at once ---
    out_gemm<<<32768 / 8, 256, 0, stream>>>(h_all + SLOT, woutB, bout, out);
}
